// benchmark_28398323761490
// MI455X (gfx1250) — matrix operand order repaired and checked by reading, not compiled, not run on hardware
//
#include <hip/hip_runtime.h>
#include <stdint.h>

typedef _Float16 h16;
typedef __attribute__((ext_vector_type(16))) _Float16 v16h;
typedef __attribute__((ext_vector_type(8)))  _Float16 v8h;
typedef __attribute__((ext_vector_type(8)))  float    v8f;
typedef __attribute__((ext_vector_type(4))) unsigned int u32x4;
typedef __attribute__((ext_vector_type(8))) int i32x8;
typedef __attribute__((ext_vector_type(4))) int i32x4;

#define LSEQ 2048
#define DMODEL 256

#if defined(__has_builtin)
#if __has_builtin(__builtin_amdgcn_tensor_load_to_lds) && __has_builtin(__builtin_amdgcn_s_wait_tensorcnt)
#define HAVE_TDM 1
#endif
#if __has_builtin(__builtin_amdgcn_permlane16)
#define HAVE_PERMLANE 1
#endif
#endif

// ---------------------------------------------------------------------------
// WMMA fragment loaders (wave32, 16x16x32 f16; layouts per CDNA5 ISA 7.12.2)
// ---------------------------------------------------------------------------
__device__ __forceinline__ v16h load_fragA(const h16* __restrict__ base, int ld,
                                           int m0, int k0) {
  const int lane = threadIdx.x & 31;
  const h16* p = base + (size_t)(m0 + (lane & 15)) * ld + k0 + ((lane >> 4) << 3);
  v8h lo = *reinterpret_cast<const v8h*>(p);
  v8h hi = *reinterpret_cast<const v8h*>(p + 16);
  v16h a;
#pragma unroll
  for (int i = 0; i < 8; ++i) { a[i] = lo[i]; a[8 + i] = hi[i]; }
  return a;
}

__device__ __forceinline__ v16h load_fragB(const h16* __restrict__ base, int ld,
                                           int n0, int k0) {
  const int lane = threadIdx.x & 31;
  const h16* p = base + (size_t)(n0 + (lane & 15)) * ld + k0 + ((lane >> 4) << 3);
  v8h lo = *reinterpret_cast<const v8h*>(p);
  v8h hi = *reinterpret_cast<const v8h*>(p + 16);
  v16h b;
#pragma unroll
  for (int i = 0; i < 8; ++i) { b[i] = lo[i]; b[8 + i] = hi[i]; }
  return b;
}

__device__ __forceinline__ v8f wmma_f16(v16h a, v16h b, v8f c) {
  return __builtin_amdgcn_wmma_f32_16x16x32_f16(false, a, false, b, (short)0, c,
                                                false, false);
}

// 16-lane max reduction (rows live across 16 lanes of a wave half).
// v_permlane16_b32 (VALU) when available; ds_bpermute fallback via __shfl_xor.
#if HAVE_PERMLANE
__device__ __forceinline__ float permxor16(float x, unsigned s0, unsigned s1) {
  unsigned u = __builtin_bit_cast(unsigned, x);
  unsigned r = __builtin_amdgcn_permlane16(u, u, s0, s1, false, false);
  return __builtin_bit_cast(float, r);
}
#endif
__device__ __forceinline__ float redmax16(float v) {
#if HAVE_PERMLANE
  v = fmaxf(v, permxor16(v, 0x67452301u, 0xEFCDAB89u));  // xor 1
  v = fmaxf(v, permxor16(v, 0x45670123u, 0xCDEF89ABu));  // xor 2
  v = fmaxf(v, permxor16(v, 0x32107654u, 0xBA98FEDCu));  // xor 4
  v = fmaxf(v, permxor16(v, 0xFEDCBA98u, 0x76543210u));  // xor 8
#else
#pragma unroll
  for (int off = 1; off < 16; off <<= 1) v = fmaxf(v, __shfl_xor(v, off, 32));
#endif
  return v;
}

// ---------------------------------------------------------------------------
__global__ void cvt_f32_f16(const float* __restrict__ s, h16* __restrict__ d, int n) {
  int i = blockIdx.x * 256 + threadIdx.x;
  if (i < n) d[i] = (h16)s[i];
}

// ---------------------------------------------------------------------------
// 16-row-block WMMA GEMM: out = oscale*epilogue(X[M,K] @ W[N,K]^T + bias)
// A panel (16 x K f16) staged to LDS via TDM tensor_load_to_lds when available.
// LDS pool reused for the LayerNorm epilogue after a barrier.
// ---------------------------------------------------------------------------
template <bool HAS_LN, bool AFFINE, bool LRELU, bool RESID>
__global__ __launch_bounds__(128) void gemm16(
    const h16* __restrict__ X, const h16* __restrict__ W,
    const float* __restrict__ bias, const float* __restrict__ gw,
    const float* __restrict__ gb, const float* __restrict__ resid,
    const float* __restrict__ alphaP, const float* __restrict__ betaP,
    float* __restrict__ outF, h16* __restrict__ outH,
    int M, int N, int K, int G, int ldF, int ldH, float oscale) {
  __shared__ __align__(16) unsigned char spool[16 * 1536 * 2];  // 48KB
  __shared__ float smean[256];
  __shared__ float srstd[256];
  float* sm = (float*)spool;  // epilogue scratch (16*N f32 <= 32KB)
  const int tid = threadIdx.x;
  const int wave = tid >> 5, lane = tid & 31;
  const int half = lane >> 4, nn = lane & 15;
  const int m0 = blockIdx.x << 4;
  const int ntiles = N >> 6;
  const int nbase = wave * (N >> 2);

#if HAVE_TDM
  h16* smA = (h16*)spool;
  if (wave == 0) {
    unsigned ldsOff = (unsigned)(uintptr_t)smA;  // LDS aperture: low 32 bits
    unsigned long long ga = (unsigned long long)(uintptr_t)(X + (size_t)m0 * K);
    unsigned mrem = (unsigned)(M - m0);
    u32x4 g0;
    g0[0] = 1u;                                   // count=1, user D#
    g0[1] = ldsOff;                               // lds_addr
    g0[2] = (unsigned)ga;                         // global_addr[31:0]
    g0[3] = (unsigned)((ga >> 32) & 0x01FFFFFFull) | 0x80000000u;  // type=2
    i32x8 g1;
    g1[0] = 0x00010000;                           // data_size = 2B, no mask
    g1[1] = (int)(((unsigned)K & 0xFFFFu) << 16); // tensor_dim0 lo
    g1[2] = (int)(((unsigned)K >> 16) | ((mrem & 0xFFFFu) << 16));
    g1[3] = (int)((mrem >> 16) | (((unsigned)K & 0xFFFFu) << 16)); // tile_dim0=K
    g1[4] = 16;                                   // tile_dim1 = 16 rows
    g1[5] = K;                                    // tensor_dim0_stride lo
    g1[6] = 0;
    g1[7] = 0;
    i32x4 gz4;
    gz4[0] = 0; gz4[1] = 0; gz4[2] = 0; gz4[3] = 0;
    i32x8 gz8;
#pragma unroll
    for (int i = 0; i < 8; ++i) gz8[i] = 0;
    __builtin_amdgcn_tensor_load_to_lds(g0, g1, gz4, gz4, gz8, 0);
    __builtin_amdgcn_s_wait_tensorcnt(0);
  }
  __syncthreads();
  const h16* Asrc = smA;
  const int am0 = 0;
#else
  const h16* Asrc = X;
  const int am0 = m0;
#endif

  v8f acc[8];
#pragma unroll
  for (int t = 0; t < 8; ++t)
#pragma unroll
    for (int r = 0; r < 8; ++r) acc[t][r] = 0.f;

  for (int k0 = 0; k0 < K; k0 += 32) {
    v16h a = load_fragA(Asrc, K, am0, k0);
    for (int t = 0; t < ntiles; ++t) {
      v16h b = load_fragB(W, K, nbase + (t << 4), k0);
      acc[t] = wmma_f16(a, b, acc[t]);
    }
  }
  __syncthreads();  // A panel no longer needed; spool becomes epilogue scratch

  for (int t = 0; t < ntiles; ++t) {
    int col = nbase + (t << 4) + nn;
    float bv = bias[col];
#pragma unroll
    for (int r = 0; r < 8; ++r) sm[(half * 8 + r) * N + col] = acc[t][r] + bv;
  }
  __syncthreads();

  if (RESID) {
    float al = alphaP[0], be = betaP[0];
    for (int i = tid; i < 16 * N; i += 128) {
      int row = i / N, col = i - row * N;
      sm[i] = al * resid[(size_t)(m0 + row) * N + col] + be * sm[i];
    }
    __syncthreads();
  }

  const int ng = N / G;
  if (HAS_LN) {
    if (tid < 16) {
      for (int grp = 0; grp < ng; ++grp) {
        const float* p = &sm[tid * N + grp * G];
        float s = 0.f, s2 = 0.f;
        for (int j = 0; j < G; ++j) { float v = p[j]; s += v; s2 += v * v; }
        float mu = s / (float)G;
        float var = s2 / (float)G - mu * mu;
        smean[tid * ng + grp] = mu;
        srstd[tid * ng + grp] = rsqrtf(var + 1e-5f);
      }
    }
    __syncthreads();
  }

  for (int i = tid; i < 16 * N; i += 128) {
    int row = i / N, col = i - row * N;
    float v = sm[i];
    if (HAS_LN) {
      int grp = col / G;
      v = (v - smean[row * ng + grp]) * srstd[row * ng + grp];
      if (AFFINE) v = v * gw[col] + gb[col];
    }
    if (LRELU) v = v > 0.f ? v : 0.01f * v;
    v *= oscale;
    if (outF) outF[(size_t)(m0 + row) * ldF + col] = v;
    if (outH) outH[(size_t)(m0 + row) * ldH + col] = (h16)v;
  }
}

// ---------------------------------------------------------------------------
// KNN: 16 sequential argmin extractions (sorted => top-8 is a prefix).
// ---------------------------------------------------------------------------
__global__ __launch_bounds__(256) void knn_kernel(const float* __restrict__ pos,
                                                  int* __restrict__ idx) {
  __shared__ float d2[LSEQ];
  __shared__ float sval[256];
  __shared__ int sidx[256];
  const int tid = threadIdx.x;
  const int row = blockIdx.x;
  float px = pos[row * 3 + 0], py = pos[row * 3 + 1], pz = pos[row * 3 + 2];
  for (int j = tid; j < LSEQ; j += 256) {
    float dx = pos[j * 3 + 0] - px;
    float dy = pos[j * 3 + 1] - py;
    float dz = pos[j * 3 + 2] - pz;
    d2[j] = dx * dx + dy * dy + dz * dz;
  }
  __syncthreads();
  for (int t = 0; t < 16; ++t) {
    float best = 3.0e38f;
    int bi = 0;
    for (int j = tid; j < LSEQ; j += 256)
      if (d2[j] < best) { best = d2[j]; bi = j; }
    sval[tid] = best; sidx[tid] = bi;
    __syncthreads();
    for (int s = 128; s > 0; s >>= 1) {
      if (tid < s && sval[tid + s] < sval[tid]) {
        sval[tid] = sval[tid + s]; sidx[tid] = sidx[tid + s];
      }
      __syncthreads();
    }
    if (tid == 0) { idx[row * 16 + t] = sidx[0]; d2[sidx[0]] = 3.0e38f; }
    __syncthreads();
  }
}

// ---------------------------------------------------------------------------
__global__ __launch_bounds__(256) void transpose16(const h16* __restrict__ src,
                                                   h16* __restrict__ dst,
                                                   int rows, int cols) {
  __shared__ h16 tile[16][17];
  int tx = threadIdx.x & 15, ty = threadIdx.x >> 4;
  int r0 = blockIdx.y << 4, c0 = blockIdx.x << 4;
  tile[ty][tx] = src[(size_t)(r0 + ty) * cols + c0 + tx];
  __syncthreads();
  dst[(size_t)(c0 + ty) * rows + r0 + tx] = tile[tx][ty];
}

// ---------------------------------------------------------------------------
// Flash attention, one wave per (16-query block, head). Q is pre-scaled.
// Dense half: 64-key chunks, 4 score WMMA + 2 rowsum WMMA (ones trick) +
// 4 P.V WMMA; only the row-max needs a cross-lane tree (permlane16).
// Sparse half: <=16 KNN keys/row, lane-per-key dot products.
// ---------------------------------------------------------------------------
__global__ __launch_bounds__(32) void attn_kernel(
    const h16* __restrict__ Q,   // [L, 256] (pre-scaled by 1/sqrt(32))
    const h16* __restrict__ Kp,  // [2L, 256]
    const h16* __restrict__ Vt,  // [256, 2L]
    const h16* __restrict__ Vh,  // [2L, 256] for sparse part
    const int* __restrict__ knn, // [L, 16]
    int kcount, int knnBase, int fullBase,
    float* __restrict__ out) {   // [L, 256]
  __shared__ h16 pl[16][64];
  const int lane = threadIdx.x;
  const int half = lane >> 4, nn = lane & 15;
  const int m0 = blockIdx.x << 4;
  const int h = blockIdx.y;

  v16h aQ = load_fragA(Q, DMODEL, m0, h * 32);
  v16h ones;
#pragma unroll
  for (int i = 0; i < 16; ++i) ones[i] = (h16)1.0f;

  float rm[8];
  v8f rl, o0, o1;
#pragma unroll
  for (int r = 0; r < 8; ++r) { rm[r] = -3.0e38f; rl[r] = 0.f; o0[r] = 0.f; o1[r] = 0.f; }

  // ---- sparse KNN keys: lane nn handles key slot nn, half handles 16 dims ----
  for (int j = 0; j < 16; ++j) {
    int qrow = m0 + j;
    float partial = 0.f;
    if (nn < kcount) {
      int key = knnBase + knn[qrow * 16 + nn];
      const h16* qp = Q + (size_t)qrow * DMODEL + h * 32 + half * 16;
      const h16* kp = Kp + (size_t)key * DMODEL + h * 32 + half * 16;
#pragma unroll
      for (int i = 0; i < 16; ++i) partial += (float)qp[i] * (float)kp[i];
    }
    float sdot = partial + __shfl_xor(partial, 16, 32);  // lane nn: score of key nn
    int r = j & 7;
    bool own = (half == (j >> 3));
    for (int t = 0; t < kcount; ++t) {
      float st = __shfl(sdot, t, 32);
      int key = knnBase + knn[qrow * 16 + t];  // uniform -> scalar load
      if (own) {
        float mn = fmaxf(rm[r], st);
        float corr = __expf(rm[r] - mn);
        float p = __expf(st - mn);
        rl[r] = rl[r] * corr + p;
        o0[r] = o0[r] * corr + p * (float)Vh[(size_t)key * DMODEL + h * 32 + nn];
        o1[r] = o1[r] * corr + p * (float)Vh[(size_t)key * DMODEL + h * 32 + 16 + nn];
        rm[r] = mn;
      }
    }
  }

  // ---- dense half: stream 64 keys per iteration ----
  for (int kb = 0; kb < LSEQ; kb += 64) {
    const int key0 = fullBase + kb;
    v8f s[4];
#pragma unroll
    for (int t = 0; t < 4; ++t) {
      v16h b = load_fragB(Kp, DMODEL, key0 + 16 * t, h * 32);
      v8f z;
#pragma unroll
      for (int r = 0; r < 8; ++r) z[r] = 0.f;
      s[t] = wmma_f16(aQ, b, z);
    }
    float mx[8];
#pragma unroll
    for (int r = 0; r < 8; ++r)
      mx[r] = fmaxf(fmaxf(s[0][r], s[1][r]), fmaxf(s[2][r], s[3][r]));
#pragma unroll
    for (int r = 0; r < 8; ++r) mx[r] = redmax16(mx[r]);
#pragma unroll
    for (int r = 0; r < 8; ++r) {
      float mn = fmaxf(rm[r], mx[r]);
      float corr = __expf(rm[r] - mn);
      rm[r] = mn;
      rl[r] *= corr; o0[r] *= corr; o1[r] *= corr;
    }
#pragma unroll
    for (int t = 0; t < 4; ++t)
#pragma unroll
      for (int r = 0; r < 8; ++r) {
        float p = __expf(s[t][r] - rm[r]);
        pl[half * 8 + r][t * 16 + nn] = (h16)p;
      }
    asm volatile("s_wait_dscnt 0" ::: "memory");  // cross-lane LDS RAW
    v16h aP0 = load_fragA(&pl[0][0], 64, 0, 0);
    v16h aP1 = load_fragA(&pl[0][0], 64, 0, 32);
    rl = wmma_f16(aP0, ones, rl);          // row-sum via WMMA (broadcast in C)
    rl = wmma_f16(aP1, ones, rl);
    v16h bv;
    bv = load_fragB(Vt, 2 * LSEQ, h * 32, key0);
    o0 = wmma_f16(aP0, bv, o0);
    bv = load_fragB(Vt, 2 * LSEQ, h * 32, key0 + 32);
    o0 = wmma_f16(aP1, bv, o0);
    bv = load_fragB(Vt, 2 * LSEQ, h * 32 + 16, key0);
    o1 = wmma_f16(aP0, bv, o1);
    bv = load_fragB(Vt, 2 * LSEQ, h * 32 + 16, key0 + 32);
    o1 = wmma_f16(aP1, bv, o1);
  }

#pragma unroll
  for (int r = 0; r < 8; ++r) {
    int row = m0 + half * 8 + r;
    float inv = 1.f / rl[r];
    out[(size_t)row * DMODEL + h * 32 + nn] = o0[r] * inv;
    out[(size_t)row * DMODEL + h * 32 + 16 + nn] = o1[r] * inv;
  }
}

// ---------------------------------------------------------------------------
__global__ void combine_scales(const float* __restrict__ a,
                               const float* __restrict__ b,
                               const float* __restrict__ sw,
                               h16* __restrict__ out, int n) {
  int i = blockIdx.x * 256 + threadIdx.x;
  if (i >= n) return;
  float s0 = sw[0], s1 = sw[1];
  float m = fmaxf(s0, s1);
  float e0 = __expf(s0 - m), e1 = __expf(s1 - m);
  float inv = 1.f / (e0 + e1);
  out[i] = (h16)((e0 * inv) * a[i] + (e1 * inv) * b[i]);
}

// ---------------------------------------------------------------------------
extern "C" void kernel_launch(void* const* d_in, const int* in_sizes, int n_in,
                              void* d_out, int out_size, void* d_ws, size_t ws_size,
                              hipStream_t stream) {
  const float* geo_feat = (const float*)d_in[0];
  const float* sem_feat = (const float*)d_in[1];
  const float* rsa_feat = (const float*)d_in[2];
  const float* pos      = (const float*)d_in[3];
  const float* geo_W = (const float*)d_in[4];
  const float* geo_b = (const float*)d_in[5];
  const float* geo_g = (const float*)d_in[6];
  const float* geo_bb = (const float*)d_in[7];
  const float* sem_W = (const float*)d_in[8];
  const float* sem_b = (const float*)d_in[9];
  const float* sem_g = (const float*)d_in[10];
  const float* sem_bb = (const float*)d_in[11];
  const float* rsa_W = (const float*)d_in[12];
  const float* rsa_b = (const float*)d_in[13];
  const float* rsa_g = (const float*)d_in[14];
  const float* rsa_bb = (const float*)d_in[15];
  const float* qg_W = (const float*)d_in[16];
  const float* qg_b = (const float*)d_in[17];
  const float* qs_W = (const float*)d_in[18];
  const float* qs_b = (const float*)d_in[19];
  const float* k_W = (const float*)d_in[20];
  const float* k_b = (const float*)d_in[21];
  const float* v_W = (const float*)d_in[22];
  const float* v_b = (const float*)d_in[23];
  const float* o_W = (const float*)d_in[24];
  const float* o_b = (const float*)d_in[25];
  const float* sw_geo = (const float*)d_in[26];
  const float* sw_sem = (const float*)d_in[27];
  const float* alpha_geo = (const float*)d_in[28];
  const float* beta_geo = (const float*)d_in[29];
  const float* alpha_sem = (const float*)d_in[30];
  const float* beta_sem = (const float*)d_in[31];
  const float* ln1_g = (const float*)d_in[32];
  const float* ln1_b = (const float*)d_in[33];
  const float* ln2_g = (const float*)d_in[34];
  const float* ln2_b = (const float*)d_in[35];
  const float* rt_W = (const float*)d_in[36];
  const float* rt_b = (const float*)d_in[37];
  const float* rt_g = (const float*)d_in[38];
  const float* rt_bb = (const float*)d_in[39];
  const float* f1_W = (const float*)d_in[40];
  const float* f1_b = (const float*)d_in[41];
  const float* f1_g = (const float*)d_in[42];
  const float* f1_bb = (const float*)d_in[43];
  const float* f2_W = (const float*)d_in[44];
  const float* f2_b = (const float*)d_in[45];
  const float* f2_g = (const float*)d_in[46];
  const float* f2_bb = (const float*)d_in[47];

  char* ws = (char*)d_ws;
  size_t off = 0;
  auto alloc = [&](size_t bytes) {
    void* p = ws + off;
    off = (off + bytes + 255) & ~(size_t)255;
    return p;
  };

  const int L = LSEQ, D = DMODEL;
  h16* geo16  = (h16*)alloc((size_t)L * 1536 * 2);
  h16* sem16  = (h16*)alloc((size_t)L * 512 * 2);
  h16* rsa16  = (h16*)alloc((size_t)L * 64 * 2);
  h16* wGeo16 = (h16*)alloc((size_t)D * 1536 * 2);
  h16* wSem16 = (h16*)alloc((size_t)D * 512 * 2);
  h16* wRsa16 = (h16*)alloc((size_t)D * 64 * 2);
  h16* wQg16  = (h16*)alloc((size_t)D * D * 2);
  h16* wQs16  = (h16*)alloc((size_t)D * D * 2);
  h16* wK16   = (h16*)alloc((size_t)D * D * 2);
  h16* wV16   = (h16*)alloc((size_t)D * D * 2);
  h16* wO16   = (h16*)alloc((size_t)D * D * 2);
  h16* wRt16  = (h16*)alloc((size_t)D * D * 2);
  h16* wF1_16 = (h16*)alloc((size_t)512 * 768 * 2);
  h16* wF2_16 = (h16*)alloc((size_t)D * 512 * 2);

  float* geoP32 = (float*)alloc((size_t)L * D * 4);
  float* semP32 = (float*)alloc((size_t)L * D * 4);
  h16* joint16 = (h16*)alloc((size_t)2 * L * D * 2);
  h16* rsaP16  = (h16*)alloc((size_t)L * D * 2);
  h16* K16     = (h16*)alloc((size_t)2 * L * D * 2);
  h16* V16     = (h16*)alloc((size_t)2 * L * D * 2);
  h16* Vt16    = (h16*)alloc((size_t)D * 2 * L * 2);
  h16* Qg16    = (h16*)alloc((size_t)L * D * 2);
  h16* Qs16    = (h16*)alloc((size_t)L * D * 2);
  int* knnIdx  = (int*)alloc((size_t)L * 16 * 4);
  float* aG8   = (float*)alloc((size_t)L * D * 4);
  float* aG16  = (float*)alloc((size_t)L * D * 4);
  float* aS8   = (float*)alloc((size_t)L * D * 4);
  float* aS16  = (float*)alloc((size_t)L * D * 4);
  h16* combG16 = (h16*)alloc((size_t)L * D * 2);
  h16* combS16 = (h16*)alloc((size_t)L * D * 2);
  h16* cat16   = (h16*)alloc((size_t)L * 768 * 2);
  h16* hid16   = (h16*)alloc((size_t)L * 512 * 2);
  (void)ws_size;

  auto cvt = [&](const float* s, h16* d, int n) {
    cvt_f32_f16<<<(n + 255) / 256, 256, 0, stream>>>(s, d, n);
  };
  cvt(geo_feat, geo16, L * 1536);
  cvt(sem_feat, sem16, L * 512);
  cvt(rsa_feat, rsa16, L * 64);
  cvt(geo_W, wGeo16, D * 1536);
  cvt(sem_W, wSem16, D * 512);
  cvt(rsa_W, wRsa16, D * 64);
  cvt(qg_W, wQg16, D * D);
  cvt(qs_W, wQs16, D * D);
  cvt(k_W, wK16, D * D);
  cvt(v_W, wV16, D * D);
  cvt(o_W, wO16, D * D);
  cvt(rt_W, wRt16, D * D);
  cvt(f1_W, wF1_16, 512 * 768);
  cvt(f2_W, wF2_16, D * 512);

  knn_kernel<<<L, 256, 0, stream>>>(pos, knnIdx);

  const float QSCALE = 0.17677669529663687f;  // 1/sqrt(32)

  gemm16<true, true, false, false><<<L / 16, 128, 0, stream>>>(
      geo16, wGeo16, geo_b, geo_g, geo_bb, nullptr, nullptr, nullptr,
      geoP32, joint16, L, D, 1536, D, D, D, 1.0f);
  gemm16<true, true, false, false><<<L / 16, 128, 0, stream>>>(
      sem16, wSem16, sem_b, sem_g, sem_bb, nullptr, nullptr, nullptr,
      semP32, joint16 + (size_t)L * D, L, D, 512, D, D, D, 1.0f);
  gemm16<true, true, false, false><<<L / 16, 128, 0, stream>>>(
      rsa16, wRsa16, rsa_b, rsa_g, rsa_bb, nullptr, nullptr, nullptr,
      nullptr, rsaP16, L, D, 64, D, D, D, 1.0f);

  gemm16<true, false, false, false><<<2 * L / 16, 128, 0, stream>>>(
      joint16, wK16, k_b, nullptr, nullptr, nullptr, nullptr, nullptr,
      nullptr, K16, 2 * L, D, D, 32, D, D, 1.0f);
  gemm16<true, false, false, false><<<2 * L / 16, 128, 0, stream>>>(
      joint16, wV16, v_b, nullptr, nullptr, nullptr, nullptr, nullptr,
      nullptr, V16, 2 * L, D, D, 32, D, D, 1.0f);

  gemm16<false, false, false, false><<<L / 16, 128, 0, stream>>>(
      joint16, wQg16, qg_b, nullptr, nullptr, nullptr, nullptr, nullptr,
      nullptr, Qg16, L, D, D, D, D, D, QSCALE);
  gemm16<false, false, false, false><<<L / 16, 128, 0, stream>>>(
      joint16 + (size_t)L * D, wQs16, qs_b, nullptr, nullptr, nullptr, nullptr,
      nullptr, nullptr, Qs16, L, D, D, D, D, D, QSCALE);

  transpose16<<<dim3(D / 16, 2 * L / 16), 256, 0, stream>>>(V16, Vt16, 2 * L, D);

  dim3 agrid(L / 16, 8);
  attn_kernel<<<agrid, 32, 0, stream>>>(Qg16, K16, Vt16, V16, knnIdx, 8, 0, L, aG8);
  attn_kernel<<<agrid, 32, 0, stream>>>(Qg16, K16, Vt16, V16, knnIdx, 16, 0, L, aG16);
  attn_kernel<<<agrid, 32, 0, stream>>>(Qs16, K16, Vt16, V16, knnIdx, 8, L, 0, aS8);
  attn_kernel<<<agrid, 32, 0, stream>>>(Qs16, K16, Vt16, V16, knnIdx, 16, L, 0, aS16);

  combine_scales<<<(L * D + 255) / 256, 256, 0, stream>>>(aG8, aG16, sw_geo, combG16, L * D);
  combine_scales<<<(L * D + 255) / 256, 256, 0, stream>>>(aS8, aS16, sw_sem, combS16, L * D);

  gemm16<true, true, false, true><<<L / 16, 128, 0, stream>>>(
      combG16, wO16, o_b, ln1_g, ln1_b, geoP32, alpha_geo, beta_geo,
      nullptr, cat16, L, D, D, D, D, 768, 1.0f);
  gemm16<true, true, false, true><<<L / 16, 128, 0, stream>>>(
      combS16, wO16, o_b, ln2_g, ln2_b, semP32, alpha_sem, beta_sem,
      nullptr, cat16 + D, L, D, D, D, D, 768, 1.0f);

  gemm16<true, true, true, false><<<L / 16, 128, 0, stream>>>(
      rsaP16, wRt16, rt_b, rt_g, rt_bb, nullptr, nullptr, nullptr,
      nullptr, cat16 + 2 * D, L, D, D, D, D, 768, 1.0f);

  gemm16<true, true, true, false><<<L / 16, 128, 0, stream>>>(
      cat16, wF1_16, f1_b, f1_g, f1_bb, nullptr, nullptr, nullptr,
      nullptr, hid16, L, 512, 768, 512, 512, 512, 1.0f);
  gemm16<true, true, true, false><<<L / 16, 128, 0, stream>>>(
      hid16, wF2_16, f2_b, f2_g, f2_bb, nullptr, nullptr, nullptr,
      (float*)d_out, nullptr, L, D, 512, D, D, D, 1.0f);
  (void)in_sizes; (void)n_in; (void)out_size;
}
